// RGCN_3358664425857
// MI455X (gfx1250) — matrix operand order repaired and checked by reading, not compiled, not run on hardware
//
#include <hip/hip_runtime.h>

typedef __attribute__((ext_vector_type(16))) __bf16 v16bf;
typedef __attribute__((ext_vector_type(8)))  __bf16 v8bf;
typedef __attribute__((ext_vector_type(8)))  float  v8f;

#define NNODES 100000
#define NEDGES 1600000
#define NRELS  8
#define KTILES 18          // K = 64 + 8*64 = 576 = 18*32 (both layers)

__device__ __forceinline__ unsigned short f2bf_bits(float f) {
  __bf16 b = (__bf16)f;                    // let clang pick native cvt on gfx1250
  return __builtin_bit_cast(unsigned short, b);
}
__device__ __forceinline__ unsigned packbf2(float lo, float hi) {
  return ((unsigned)f2bf_bits(hi) << 16) | (unsigned)f2bf_bits(lo);
}

// ---------- count edges per (dst, rel) segment -------------------------
__global__ void k_count(const int* __restrict__ dst, const int* __restrict__ et,
                        int* __restrict__ cnt) {
  int e = blockIdx.x * blockDim.x + threadIdx.x;
  if (e < NEDGES) atomicAdd(&cnt[dst[e] * NRELS + et[e]], 1);
}

// ---------- f32 -> bf16 feature conversion -----------------------------
__global__ void k_cvt(const float* __restrict__ in, unsigned short* __restrict__ out, int n) {
  int i = blockIdx.x * blockDim.x + threadIdx.x;
  if (i < n) out[i] = f2bf_bits(in[i]);
}

// ---------- scatter: Agg[dst*8+rel] += featbf[src]*norm  (pk bf16 atomics)
// one wave32 per edge; each lane handles one packed pair (2 feats)
__global__ void k_scatter(const unsigned* __restrict__ featbf,  // [N][32] packed 2xbf16
                          const int* __restrict__ src, const int* __restrict__ dst,
                          const int* __restrict__ et, const int* __restrict__ cnt,
                          unsigned* __restrict__ Agg) {         // [N*8][32] packed 2xbf16
  unsigned t = blockIdx.x * blockDim.x + threadIdx.x;
  unsigned e = t >> 5;
  unsigned lane = t & 31u;
  if (e >= NEDGES) return;
  int s = src[e], d = dst[e], r = et[e];
  int seg = d * NRELS + r;
  int c = cnt[seg];
  float norm = 1.0f / (float)(c > 0 ? c : 1);
  unsigned pk = featbf[(size_t)s * 32 + lane];
  float lo = __builtin_bit_cast(float, pk << 16) * norm;          // exact bf16->f32
  float hi = __builtin_bit_cast(float, pk & 0xffff0000u) * norm;
  unsigned dpk = packbf2(lo, hi);
  unsigned long long addr = (unsigned long long)(Agg + (size_t)seg * 32 + lane);
  asm volatile("global_atomic_pk_add_bf16 %0, %1, off"
               :: "v"(addr), "v"(dpk) : "memory");
}

// ---------- pre-swizzle Wcat = [root; W_0..W_7] into WMMA B-frag layout -
// B (32x16 bf16): j<8 -> K = h*8+j ; j>=8 -> K = 16+h*8+(j-8)  (h = lane>>4)
template<int NC>
__global__ void k_prep_w(const float* __restrict__ root, const float* __restrict__ W,
                         unsigned short* __restrict__ out) {
  constexpr int NT = NC / 16;
  int t = blockIdx.x * blockDim.x + threadIdx.x;
  if (t >= KTILES * NT * 32 * 16) return;
  int j    = t & 15;
  int lane = (t >> 4) & 31;
  int rest = t >> 9;
  int nt = rest % NT;
  int kt = rest / NT;
  int k = kt * 32 + (j >> 3) * 16 + (lane >> 4) * 8 + (j & 7);
  int n = nt * 16 + (lane & 15);
  float v;
  if (k < 64) {
    v = root[k * NC + n];
  } else {
    int r  = (k - 64) >> 6;
    int kk = (k - 64) & 63;
    v = W[(r * 64 + kk) * NC + n];
  }
  out[t] = f2bf_bits(v);
}

// ---------- WMMA GEMM: out[N, NT*16] = [Xbf | Agg] (N x 576) @ Bfrag ----
template<int NT, bool RELU, bool BF16OUT>
__global__ void k_gemm(const unsigned short* __restrict__ X,   // [N][64] bf16
                       const unsigned short* __restrict__ A,   // [N][512] bf16
                       const unsigned short* __restrict__ Bf,
                       const float* __restrict__ bias, void* __restrict__ outv) {
  constexpr int NCOLS = NT * 16;
  int wave = blockIdx.x * (blockDim.x >> 5) + (threadIdx.x >> 5);
  int lane = threadIdx.x & 31;
  int row0 = wave * 16;
  if (row0 >= NNODES) return;   // whole-wave exit: EXEC all-1s at every WMMA
  int h = lane >> 4;
  int m = lane & 15;
  const unsigned short* xrow = X + (size_t)(row0 + m) * 64;    // K =  0..63
  const unsigned short* arow = A + (size_t)(row0 + m) * 512;   // K = 64..575

  v8f acc[NT] = {};
  #pragma unroll
  for (int kt = 0; kt < KTILES; ++kt) {
    const unsigned short* base = (kt < 2) ? (xrow + kt * 32) : (arow + (kt - 2) * 32);
    // A-frag (16-bit A 16x32 layout): j<8 -> K=h*8+j ; j>=8 -> K=16+h*8+(j-8)
    v8bf c0 = *(const v8bf*)(base + h * 8);
    v8bf c1 = *(const v8bf*)(base + 16 + h * 8);
    v16bf a = __builtin_shufflevector(c0, c1, 0,1,2,3,4,5,6,7,8,9,10,11,12,13,14,15);
    #pragma unroll
    for (int nt = 0; nt < NT; ++nt) {
      v16bf b = *(const v16bf*)(Bf + ((size_t)(kt * NT + nt) * 32 + lane) * 16);
      acc[nt] = __builtin_amdgcn_wmma_f32_16x16x32_bf16(
          false, a, false, b, (short)0, acc[nt], false, false);
    }
  }
  // C/D layout: lane n = L&15, half h; c[v] = C[v + 8h][n]
  #pragma unroll
  for (int nt = 0; nt < NT; ++nt) {
    #pragma unroll
    for (int v = 0; v < 8; ++v) {
      int rr = row0 + v + 8 * h;
      int cc = nt * 16 + m;
      float rv = acc[nt][v] + bias[cc];
      if (RELU) rv = rv > 0.0f ? rv : 0.0f;
      if (BF16OUT) ((unsigned short*)outv)[(size_t)rr * NCOLS + cc] = f2bf_bits(rv);
      else         ((float*)outv)[(size_t)rr * NCOLS + cc] = rv;
    }
  }
}

extern "C" void kernel_launch(void* const* d_in, const int* in_sizes, int n_in,
                              void* d_out, int out_size, void* d_ws, size_t ws_size,
                              hipStream_t stream) {
  const float* x     = (const float*)d_in[0];
  const int*   ei    = (const int*)d_in[1];
  const int*   et    = (const int*)d_in[2];
  const float* W1    = (const float*)d_in[3];
  const float* root1 = (const float*)d_in[4];
  const float* b1    = (const float*)d_in[5];
  const float* W2    = (const float*)d_in[6];
  const float* root2 = (const float*)d_in[7];
  const float* b2    = (const float*)d_in[8];
  const int* src = ei;            // edge_index[0]
  const int* dst = ei + NEDGES;   // edge_index[1]

  char* ws = (char*)d_ws;
  size_t off = 0;
  int*            cnt = (int*)(ws + off);            off += (size_t)NNODES * NRELS * 4;       // 3.2 MB
  unsigned short* Agg = (unsigned short*)(ws + off); off += (size_t)NNODES * NRELS * 64 * 2;  // 102.4 MB
  unsigned short* xbf = (unsigned short*)(ws + off); off += (size_t)NNODES * 64 * 2;          // 12.8 MB
  unsigned short* hbf = (unsigned short*)(ws + off); off += (size_t)NNODES * 64 * 2;          // 12.8 MB
  unsigned short* B1  = (unsigned short*)(ws + off); off += (size_t)KTILES * 4 * 32 * 16 * 2;
  unsigned short* B2  = (unsigned short*)(ws + off); off += (size_t)KTILES * 2 * 32 * 16 * 2;
  if (ws_size < off) return;

  hipMemsetAsync(cnt, 0, (size_t)NNODES * NRELS * 4, stream);
  hipMemsetAsync(Agg, 0, (size_t)NNODES * NRELS * 64 * 2, stream);

  k_count<<<(NEDGES + 255) / 256, 256, 0, stream>>>(dst, et, cnt);
  k_cvt<<<(NNODES * 64 + 255) / 256, 256, 0, stream>>>(x, xbf, NNODES * 64);
  k_prep_w<64><<<(KTILES * 4 * 512 + 255) / 256, 256, 0, stream>>>(root1, W1, B1);
  k_prep_w<32><<<(KTILES * 2 * 512 + 255) / 256, 256, 0, stream>>>(root2, W2, B2);

  unsigned sthreads = (unsigned)NEDGES * 32u;
  k_scatter<<<(sthreads + 255) / 256, 256, 0, stream>>>(
      (const unsigned*)xbf, src, dst, et, cnt, (unsigned*)Agg);

  int waves  = NNODES / 16;       // 6250, exact
  int blocks = (waves + 7) / 8;   // 8 waves per 256-thread block
  k_gemm<4, true, true><<<blocks, 256, 0, stream>>>(xbf, Agg, B1, b1, hbf);

  hipMemsetAsync(Agg, 0, (size_t)NNODES * NRELS * 64 * 2, stream);
  k_scatter<<<(sthreads + 255) / 256, 256, 0, stream>>>(
      (const unsigned*)hbf, src, dst, et, cnt, (unsigned*)Agg);
  k_gemm<2, false, false><<<blocks, 256, 0, stream>>>(hbf, Agg, B2, b2, (float*)d_out);
}
